// HIERLoss_16561393894098
// MI455X (gfx1250) — compile-verified, matrix operand order checked by reading, not run on hardware
//
#include <hip/hip_runtime.h>
#include <stdint.h>

// ---- problem constants (from reference) ----
#define D        512
#define BS       1024
#define NPROX    512
#define NROWS    1536          // BS + NPROX
#define TPER     50
#define TOPKK    30
#define SQRT_C   0.31622776601683794f
#define TAU_INV  10.0f         // 1/TAU
#define MRGIN    0.1f

// supertile = 32x32 output (2x2 register tile of 16x16 WMMA tiles)
#define STILES   48                          // 1536/32
#define NSUPER   (STILES * (STILES + 1) / 2) // upper triangle: 1176 waves

// ---- WMMA types (gfx1250, wave32) ----
typedef __attribute__((ext_vector_type(16))) __bf16 v16bf;
typedef __attribute__((ext_vector_type(8)))  float  v8f;
union FragU { v16bf v; unsigned u[8]; };

// ---- helpers ----
static __device__ __forceinline__ unsigned short f2bf(float f) {
    unsigned x = __float_as_uint(f);
    unsigned r = (x + 0x7fffu + ((x >> 16) & 1u)) >> 16;   // RNE
    return (unsigned short)r;
}
static __device__ __forceinline__ float bf2f(unsigned short h) {
    return __uint_as_float(((unsigned)h) << 16);
}
static __device__ __forceinline__ unsigned pcg(unsigned x) {
    x = x * 747796405u + 2891336453u;
    x = ((x >> ((x >> 28) + 4u)) ^ x) * 277803737u;
    return (x >> 22) ^ x;
}
static __device__ __forceinline__ float gumbel_from(unsigned h) {
    float u = (float)((h >> 8) & 0xFFFFFFu) * (1.0f / 16777216.0f);
    u = fminf(fmaxf(u, 1e-7f), 1.0f - 1e-7f);
    return -logf(-logf(u));
}

// ============================================================
// K0: zero accumulators
// ============================================================
__global__ void zero_accum_kernel(float* acc) {
    if (threadIdx.x < 8) acc[threadIdx.x] = 0.0f;
}

// ============================================================
// K1: build all_nodes rows: rows<BS copy z_s; rows>=BS apply
//     to_poincare(lcas_param). Emit bf16 hi/lo split + row |x|^2.
// ============================================================
__global__ void prep_kernel(const float* __restrict__ z,
                            const float* __restrict__ lcas,
                            unsigned short* __restrict__ Ahi,
                            unsigned short* __restrict__ Alo,
                            float* __restrict__ x2) {
    const int row = blockIdx.x;            // 0..1535
    const int tid = threadIdx.x;           // 0..255
    __shared__ float red[256];
    __shared__ float sAllSh;

    const float* src = (row < BS) ? (z + (size_t)row * D)
                                  : (lcas + (size_t)(row - BS) * D);
    float v0 = src[tid];
    float v1 = src[tid + 256];
    red[tid] = v0 * v0 + v1 * v1;
    __syncthreads();
    for (int s = 128; s > 0; s >>= 1) {
        if (tid < s) red[tid] += red[tid + s];
        __syncthreads();
    }
    if (tid == 0) {
        float tot = red[0];
        float sAll = 1.0f;
        if (row >= BS) {
            float n1 = sqrtf(tot);
            float s1 = fminf(1.0f, 2.3f / fmaxf(n1, 1e-12f));     // clip_feat
            float nu = fmaxf(n1 * s1, 1e-5f);                     // expmap0
            float sc2 = tanhf(SQRT_C * nu) / (SQRT_C * nu);
            sAll = s1 * sc2;
            float nv = fmaxf(n1 * sAll, 1e-5f);                   // project
            float maxnorm = (1.0f - 1e-3f) / SQRT_C;
            if (nv > maxnorm) sAll *= maxnorm / nv;
        }
        sAllSh = sAll;
        x2[row] = tot * sAll * sAll;
    }
    __syncthreads();
    float s = sAllSh;
    float a0 = v0 * s, a1 = v1 * s;
    unsigned short h0 = f2bf(a0), h1 = f2bf(a1);
    size_t base = (size_t)row * D;
    Ahi[base + tid]       = h0;
    Ahi[base + tid + 256] = h1;
    Alo[base + tid]       = f2bf(a0 - bf2f(h0));
    Alo[base + tid + 256] = f2bf(a1 - bf2f(h1));
}

// ============================================================
// K2: symmetric Gram via split-bf16 WMMA, 2x2 register tiling.
//     One wave = one 32x32 supertile of the UPPER triangle;
//     Poincare-distance epilogue stores value + its mirror.
// ============================================================
__global__ void dist_wmma_kernel(const unsigned short* __restrict__ Ahi,
                                 const unsigned short* __restrict__ Alo,
                                 const float* __restrict__ x2,
                                 float* __restrict__ dist) {
    const int w = (int)((blockIdx.x * blockDim.x + threadIdx.x) >> 5);
    const int lane = threadIdx.x & 31;
    if (w >= NSUPER) return;

    // map linear index -> (TM, TN) in upper triangle, TM <= TN (wave-uniform)
    int TM = 0, rem = w;
    while (rem >= STILES - TM) { rem -= STILES - TM; ++TM; }
    const int TN = TM + rem;

    const int half = lane >> 4, mrow = lane & 15;
    const size_t ar0 = (size_t)(TM * 32 + mrow) * D;        // A tile 0 rows
    const size_t ar1 = ar0 + (size_t)16 * D;                // A tile 1 rows
    const size_t br0 = (size_t)(TN * 32 + mrow) * D;        // B tile 0 rows
    const size_t br1 = br0 + (size_t)16 * D;                // B tile 1 rows

    v8f acc00 = {}, acc01 = {}, acc10 = {}, acc11 = {};
    for (int kb = 0; kb < D; kb += 32) {
        FragU a0h, a0l, a1h, a1l, b0h, b0l, b1h, b1l;
#pragma unroll
        for (int v = 0; v < 8; ++v) {
            // ISA 16-bit A-fragment layout: VGPR v <- consecutive K pair
            const int K0 = kb + ((v >> 2) << 4) + (half << 3) + ((v & 3) << 1);
            a0h.u[v] = *(const unsigned*)(Ahi + ar0 + K0);
            a0l.u[v] = *(const unsigned*)(Alo + ar0 + K0);
            a1h.u[v] = *(const unsigned*)(Ahi + ar1 + K0);
            a1l.u[v] = *(const unsigned*)(Alo + ar1 + K0);
            b0h.u[v] = *(const unsigned*)(Ahi + br0 + K0);
            b0l.u[v] = *(const unsigned*)(Alo + br0 + K0);
            b1h.u[v] = *(const unsigned*)(Ahi + br1 + K0);
            b1l.u[v] = *(const unsigned*)(Alo + br1 + K0);
        }
        if (kb + 32 < D) {
            __builtin_prefetch(Ahi + ar0 + kb + 32, 0, 3);  // global_prefetch_b8
            __builtin_prefetch(Ahi + br0 + kb + 32, 0, 3);
        }
        // xy ~= hi*hi + hi*lo + lo*hi  (split-bf16 ~ f32 accuracy)
#define WM(ACC, A, B) \
        ACC = __builtin_amdgcn_wmma_f32_16x16x32_bf16(false, (A).v, false, (B).v, \
                                                      (short)0, ACC, false, false)
        WM(acc00, a0h, b0h); WM(acc00, a0h, b0l); WM(acc00, a0l, b0h);
        WM(acc01, a0h, b1h); WM(acc01, a0h, b1l); WM(acc01, a0l, b1h);
        WM(acc10, a1h, b0h); WM(acc10, a1h, b0l); WM(acc10, a1l, b0h);
        WM(acc11, a1h, b1h); WM(acc11, a1h, b1l); WM(acc11, a1l, b1h);
#undef WM
    }

    // epilogue: Gram -> Poincare distance, write value + mirror
    auto emit = [&](v8f acc, int tm16, int tn16) {
        const int gn = tn16 * 16 + mrow;       // C/D layout: N = lane%16
        const float y2 = x2[gn];
#pragma unroll
        for (int v = 0; v < 8; ++v) {
            const int gm = tm16 * 16 + v + half * 8;  // M = v + 8*(lane/16)
            float xy = acc[v];
            float xx = x2[gm];
            float a   = 1.0f - 0.2f * xy + 0.1f * y2;          // 2C=0.2, C=0.1
            float b   = 1.0f - 0.1f * xx;
            float den = fmaxf(1.0f - 0.2f * xy + 0.01f * xx * y2, 1e-12f);
            float num2 = fmaxf(a * a * xx - 2.0f * a * b * xy + b * b * y2, 0.0f);
            float m = sqrtf(num2) / den;
            float t = fminf(fmaxf(SQRT_C * m, -1.0f + 1e-5f), 1.0f - 1e-5f);
            float dd = (2.0f / SQRT_C) * 0.5f * logf((1.0f + t) / (1.0f - t));
            dist[(size_t)gm * NROWS + gn] = dd;
            if (gm != gn) dist[(size_t)gn * NROWS + gm] = dd;   // symmetry
        }
    };
    emit(acc00, TM * 2,     TN * 2);
    emit(acc01, TM * 2,     TN * 2 + 1);
    emit(acc10, TM * 2 + 1, TN * 2);
    emit(acc11, TM * 2 + 1, TN * 2 + 1);
}

// ============================================================
// K3: per-row top-30 of sim -> nn 0/1 matrix (one block per row)
// ============================================================
__global__ void topk_kernel(const float* __restrict__ dist, int n, int rowOff,
                            const int* __restrict__ y,
                            unsigned char* __restrict__ nn) {
    const int i = blockIdx.x;
    const int tid = threadIdx.x;
    __shared__ float s[1024];
    __shared__ unsigned char sel[1024];
    __shared__ float rv[256];
    __shared__ int   ri[256];

    const float* row = dist + (size_t)(rowOff + i) * NROWS + rowOff;
    for (int j = tid; j < n; j += 256) {
        float v = expf(-row[j]);
        if (y) v += (y[i] == y[j]) ? 1.0f : 0.0f;
        s[j] = v; sel[j] = 0;
    }
    __syncthreads();
    for (int it = 0; it < TOPKK; ++it) {
        float bv = -1e30f; int bi = -1;
        for (int j = tid; j < n; j += 256) {
            float v = s[j];
            if (v > bv) { bv = v; bi = j; }
        }
        rv[tid] = bv; ri[tid] = bi;
        __syncthreads();
        for (int st = 128; st > 0; st >>= 1) {
            if (tid < st) {
                if (rv[tid + st] > rv[tid] ||
                    (rv[tid + st] == rv[tid] && ri[tid + st] < ri[tid])) {
                    rv[tid] = rv[tid + st]; ri[tid] = ri[tid + st];
                }
            }
            __syncthreads();
        }
        if (tid == 0) { int b = ri[0]; sel[b] = 1; s[b] = -1e30f; }
        __syncthreads();
    }
    for (int j = tid; j < n; j += 256) nn[(size_t)i * n + j] = sel[j];
}

// ============================================================
// K4: reciprocal-NN masks + uniform categorical triplet sampling
// ============================================================
__global__ void sample_kernel(const unsigned char* __restrict__ nn, int n,
                              unsigned char* __restrict__ valid,
                              int* __restrict__ tj, int* __restrict__ tk,
                              float* __restrict__ cntAccum, unsigned seed) {
    const int i = blockIdx.x;
    const int tid = threadIdx.x;
    __shared__ unsigned char pos[1024];
    __shared__ int red[256];
    __shared__ int cposSh;

    int local = 0;
    for (int j = tid; j < n; j += 256) {
        unsigned char p = (j != i) && nn[(size_t)i * n + j] && nn[(size_t)j * n + i];
        pos[j] = p; local += p;
    }
    red[tid] = local;
    __syncthreads();
    for (int s = 128; s > 0; s >>= 1) {
        if (tid < s) red[tid] += red[tid + s];
        __syncthreads();
    }
    if (tid == 0) {
        cposSh = red[0];
        unsigned char v = (red[0] >= 2) ? 1 : 0;
        valid[i] = v;
        if (v) atomicAdd(cntAccum, 1.0f);
    }
    __syncthreads();
    const int cpos = cposSh;
    const int nneg = n - cpos;   // neg mask = !pos (diag included, matches sym<1)
    if (tid < TPER) {
        unsigned hp = pcg(seed ^ (unsigned)(i * 1664525 + tid * 1013904223 + 0x1111));
        unsigned hn = pcg(seed ^ (unsigned)(i * 22695477 + tid * 69069 + 0x2222));
        int jsel = 0, ksel = 0;
        if (cpos > 0) {
            int tgt = (int)(hp % (unsigned)cpos), c = 0;
            for (int j = 0; j < n; ++j) { if (pos[j]) { if (c == tgt) { jsel = j; break; } ++c; } }
        } else {
            jsel = (int)(hp % (unsigned)n);
        }
        {
            int tgt = (int)(hn % (unsigned)nneg), c = 0;
            for (int j = 0; j < n; ++j) { if (!pos[j]) { if (c == tgt) { ksel = j; break; } ++c; } }
        }
        tj[(size_t)i * TPER + tid] = jsel;
        tk[(size_t)i * TPER + tid] = ksel;
    }
}

// ============================================================
// K5: gHHC forward. One wave per triplet; hard Gumbel => argmax.
// ============================================================
__global__ void ghhc_kernel(const float* __restrict__ dist,
                            const int* __restrict__ tj, const int* __restrict__ tk,
                            const unsigned char* __restrict__ valid,
                            int nTrip, int anchorOff,
                            float* __restrict__ sumAccum, unsigned seed) {
    const int gw = (int)((blockIdx.x * blockDim.x + threadIdx.x) >> 5);
    const int lane = threadIdx.x & 31;
    if (gw >= nTrip) return;
    const int i = gw / TPER;
    if (!valid[i]) return;
    const int j = tj[gw], k = tk[gw];
    const float* di = dist + (size_t)(anchorOff + i) * NROWS + BS; // cols = proxies
    const float* dj = dist + (size_t)(anchorOff + j) * NROWS + BS;
    const float* dk = dist + (size_t)(anchorOff + k) * NROWS + BS;

    float bv1 = -1e30f, bv2 = -1e30f;
    int bi1 = 0, bi2 = 0;
#pragma unroll 4
    for (int p = lane; p < NPROX; p += 32) {
        float a = di[p], b = dj[p], c = dk[p];
        float mij  = fmaxf(a, b);
        float mijk = fmaxf(mij, c);
        unsigned hb = (unsigned)(gw * 2654435761u + (unsigned)p * 40503u);
        float l1 = -mij  * TAU_INV + gumbel_from(pcg(seed ^ 0x3333u ^ hb));
        float l2 = -mijk * TAU_INV + gumbel_from(pcg(seed ^ 0x4444u ^ hb));
        if (l1 > bv1 || (l1 == bv1 && p < bi1)) { bv1 = l1; bi1 = p; }
        if (l2 > bv2 || (l2 == bv2 && p < bi2)) { bv2 = l2; bi2 = p; }
    }
    for (int off = 16; off > 0; off >>= 1) {
        float ov = __shfl_xor(bv1, off, 32); int oi = __shfl_xor(bi1, off, 32);
        if (ov > bv1 || (ov == bv1 && oi < bi1)) { bv1 = ov; bi1 = oi; }
        ov = __shfl_xor(bv2, off, 32); oi = __shfl_xor(bi2, off, 32);
        if (ov > bv2 || (ov == bv2 && oi < bi2)) { bv2 = ov; bi2 = oi; }
    }
    if (lane == 0 && bi1 != bi2) {
        float hc = fmaxf(di[bi1] - di[bi2] + MRGIN, 0.0f)
                 + fmaxf(dj[bi1] - dj[bi2] + MRGIN, 0.0f)
                 + fmaxf(dk[bi2] - dk[bi1] + MRGIN, 0.0f);
        if (hc != 0.0f) atomicAdd(sumAccum, hc);
    }
}

// ============================================================
// K6: finalize  loss = sum1/max(cnt1*50,1) + sum2/max(cnt2*50,1)
// ============================================================
__global__ void finalize_kernel(const float* __restrict__ acc, float* __restrict__ out) {
    if (blockIdx.x == 0 && threadIdx.x == 0) {
        float l1 = acc[0] / fmaxf(acc[1] * (float)TPER, 1.0f);
        float l2 = acc[2] / fmaxf(acc[3] * (float)TPER, 1.0f);
        out[0] = l1 + l2;
    }
}

// ============================================================
extern "C" void kernel_launch(void* const* d_in, const int* in_sizes, int n_in,
                              void* d_out, int out_size, void* d_ws, size_t ws_size,
                              hipStream_t stream) {
    (void)in_sizes; (void)n_in; (void)out_size; (void)ws_size;
    const float* z_s  = (const float*)d_in[0];
    const float* lcas = (const float*)d_in[1];
    const int*   y    = (const int*)d_in[2];
    float* out = (float*)d_out;

    char* ws = (char*)d_ws;
    size_t off = 0;
    auto take = [&](size_t bytes) {
        void* p = ws + off;
        off = (off + bytes + 255) & ~(size_t)255;
        return p;
    };
    unsigned short* Ahi  = (unsigned short*)take((size_t)NROWS * D * 2);
    unsigned short* Alo  = (unsigned short*)take((size_t)NROWS * D * 2);
    float*          x2   = (float*)take((size_t)NROWS * 4);
    float*          dist = (float*)take((size_t)NROWS * NROWS * 4);
    unsigned char*  nn1  = (unsigned char*)take((size_t)BS * BS);
    unsigned char*  nn2  = (unsigned char*)take((size_t)NPROX * NPROX);
    int*            tj1  = (int*)take((size_t)BS * TPER * 4);
    int*            tk1  = (int*)take((size_t)BS * TPER * 4);
    unsigned char*  vld1 = (unsigned char*)take(BS);
    int*            tj2  = (int*)take((size_t)NPROX * TPER * 4);
    int*            tk2  = (int*)take((size_t)NPROX * TPER * 4);
    unsigned char*  vld2 = (unsigned char*)take(NPROX);
    float*          acc  = (float*)take(8 * 4);   // [sum1,cnt1,sum2,cnt2,...]

    zero_accum_kernel<<<1, 64, 0, stream>>>(acc);
    prep_kernel<<<NROWS, 256, 0, stream>>>(z_s, lcas, Ahi, Alo, x2);

    // upper-triangle 32x32 supertiles, one wave each; 8 waves/block
    dist_wmma_kernel<<<(NSUPER + 7) / 8, 256, 0, stream>>>(Ahi, Alo, x2, dist);

    topk_kernel<<<BS, 256, 0, stream>>>(dist, BS, 0, y, nn1);
    topk_kernel<<<NPROX, 256, 0, stream>>>(dist, NPROX, BS, nullptr, nn2);

    sample_kernel<<<BS, 256, 0, stream>>>(nn1, BS, vld1, tj1, tk1, acc + 1, 0xA5A5A5A5u);
    sample_kernel<<<NPROX, 256, 0, stream>>>(nn2, NPROX, vld2, tj2, tk2, acc + 3, 0x5A5A5A5Au);

    const int nT1 = BS * TPER;     // 51200 waves
    const int nT2 = NPROX * TPER;  // 25600 waves
    ghhc_kernel<<<nT1 / 8, 256, 0, stream>>>(dist, tj1, tk1, vld1, nT1, 0,  acc + 0, 0xC0FFEE01u);
    ghhc_kernel<<<nT2 / 8, 256, 0, stream>>>(dist, tj2, tk2, vld2, nT2, BS, acc + 2, 0xC0FFEE02u);

    finalize_kernel<<<1, 32, 0, stream>>>(acc, out);
}
